// Backflow_10806137717330
// MI455X (gfx1250) — matrix operand order repaired and checked by reading, not compiled, not run on hardware
//
#include <hip/hip_runtime.h>

typedef __attribute__((ext_vector_type(16))) __bf16 bf16x16;
typedef __attribute__((ext_vector_type(8)))  float  f32x8;
typedef int i32x4 __attribute__((vector_size(16)));   // matches builtin param type
typedef unsigned short u16;
typedef unsigned int   u32;

union FragAB {
  bf16x16 v;
  uint4 q[2];
};

__device__ __forceinline__ u16 f32_to_bf16(float f) {
  u32 u = __float_as_uint(f);
  u32 r = 0x7FFFu + ((u >> 16) & 1u);   // round to nearest even
  u += r;
  return (u16)(u >> 16);
}

// ---------------- CDNA5 async global->LDS helpers ----------------
#if defined(__gfx1250__) && __has_builtin(__builtin_amdgcn_global_load_async_to_lds_b128)
#define HAVE_ASYNC_LDS 1
#else
#define HAVE_ASYNC_LDS 0
#endif

__device__ __forceinline__ void g2l16(const u16* __restrict__ g, u16* l) {
#if HAVE_ASYNC_LDS
  __builtin_amdgcn_global_load_async_to_lds_b128(
      (i32x4 __attribute__((address_space(1)))*)(g),
      (i32x4 __attribute__((address_space(3)))*)(l), 0, 0);
#else
  *(uint4*)l = *(const uint4*)g;     // sync fallback: global_load + ds_store
#endif
}

template <int N>
__device__ __forceinline__ void async_wait() {
#if HAVE_ASYNC_LDS
#if __has_builtin(__builtin_amdgcn_s_wait_asynccnt)
  __builtin_amdgcn_s_wait_asynccnt(N);
#elif defined(__AMDGCN__)
  asm volatile("s_wait_asynccnt %0" :: "i"(N) : "memory");
#endif
#endif
}

// ---------------- prep kernels ----------------

// onehot(x) as bf16 rows: oh[b][4*i + x[b,i]] = 1.0bf16
__global__ void onehot_kernel(const int* __restrict__ x, u16* __restrict__ oh, int total) {
  int t = blockIdx.x * blockDim.x + threadIdx.x;
  if (t >= total) return;
  int v = x[t] & 3;
  u32 w[2] = {0u, 0u};
  w[v >> 1] = 0x3F80u << ((v & 1) * 16);  // bf16 1.0 in the right 16-bit slot
  uint2 p; p.x = w[0]; p.y = w[1];
  *(uint2*)(oh + 4 * (size_t)t) = p;
}

// src[K][N] f32 (row-major) -> dst[N][K] bf16, LDS-tiled so both sides coalesce.
// grid (N/32, K/32), block (32, 8)
__global__ __launch_bounds__(256) void transpose_to_bf16(
    const float* __restrict__ src, u16* __restrict__ dst, int K, int N) {
  __shared__ u16 tile[32][33];
  const int kt = blockIdx.y * 32, nt = blockIdx.x * 32;
  const int tx = threadIdx.x, ty = threadIdx.y;
#pragma unroll
  for (int i = 0; i < 32; i += 8) {
    tile[ty + i][tx] = f32_to_bf16(src[(size_t)(kt + ty + i) * N + (nt + tx)]);
  }
  __syncthreads();
#pragma unroll
  for (int i = 0; i < 32; i += 8) {
    dst[(size_t)(nt + ty + i) * K + (kt + tx)] = tile[tx][ty + i];
  }
}

// ---------------- WMMA GEMM (LDS double-buffered) ----------------
// C[M][N] = A[M][K](bf16,row-major) @ Bt[N][K](bf16,N-major) + bias, optional relu,
// output f32 or bf16. Block = 8 waves (2x4), block tile 64x256, wave tile 32x64.
// K-step 32. Tiles staged via async global->LDS (5 async instr / wave / tile).

__device__ __forceinline__ void stage_tile(
    const u16* __restrict__ A, const u16* __restrict__ Bt,
    u16* bufA, u16* bufB, int row0, int col0, int K, int k0, int t) {
  {  // A tile 64x32: 256 chunks of 8 u16, one per thread
    const int row = t >> 2, kk = (t & 3) * 8;
    g2l16(A + (size_t)(row0 + row) * K + k0 + kk, bufA + row * 32 + kk);
  }
#pragma unroll
  for (int c = 0; c < 4; ++c) {  // B tile 256x32: 1024 chunks, 4 per thread
    const int idx = t + c * 256;
    const int n = idx >> 2, kk = (idx & 3) * 8;
    g2l16(Bt + (size_t)(col0 + n) * K + k0 + kk, bufB + n * 32 + kk);
  }
}

template <int RELU, int OUT_BF16>
__global__ __launch_bounds__(256) void gemm_bf16(
    const u16* __restrict__ A, const u16* __restrict__ Bt,
    const float* __restrict__ bias, void* __restrict__ Cout,
    int M, int N, int K)
{
  __shared__ u16 smem[2][64 * 32 + 256 * 32];   // 2 x 20 KiB

  const int t    = threadIdx.x;
  const int lane = t & 31;
  const int wave = t >> 5;
  const int wr = wave & 1;        // 2 wave rows
  const int wc = wave >> 1;       // 4 wave cols
  const int row0 = blockIdx.y * 64;
  const int col0 = blockIdx.x * 256;
  const int r0 = row0 + wr * 32;
  const int c0 = col0 + wc * 64;
  const int hl  = lane >> 4;      // lane half: K sub-range select per ISA layout
  const int l16 = lane & 15;

  f32x8 acc[2][4] = {};

  const int NSTEP = K / 32;
  stage_tile(A, Bt, smem[0], smem[0] + 64 * 32, row0, col0, K, 0, t);

  for (int s = 0; s < NSTEP; ++s) {
    const int cur = s & 1;
    if (s + 1 < NSTEP) {
      u16* nb = smem[cur ^ 1];
      stage_tile(A, Bt, nb, nb + 64 * 32, row0, col0, K, (s + 1) * 32, t);
      async_wait<5>();            // current tile's 5 asyncs done; next 5 in flight
    } else {
      async_wait<0>();
    }
    __syncthreads();

    const u16* bufA = smem[cur];
    const u16* bufB = smem[cur] + 64 * 32;

    FragAB a0, a1, b[4];
    // A frag (16x32): lane half 0 -> K {0..7,16..23}; half 1 -> {8..15,24..31}
    const u16* ap = bufA + (wr * 32 + l16) * 32 + hl * 8;
    a0.q[0] = *(const uint4*)(ap);
    a0.q[1] = *(const uint4*)(ap + 16);
    a1.q[0] = *(const uint4*)(ap + 16 * 32);
    a1.q[1] = *(const uint4*)(ap + 16 * 32 + 16);
#pragma unroll
    for (int j = 0; j < 4; ++j) {
      // B frag (32x16, N-major): lane half 0 -> K {0..7,16..23}; half 1 -> {8..15,24..31}
      const u16* bp = bufB + (wc * 64 + j * 16 + l16) * 32 + hl * 8;
      b[j].q[0] = *(const uint4*)(bp);
      b[j].q[1] = *(const uint4*)(bp + 16);
    }
#pragma unroll
    for (int j = 0; j < 4; ++j) {
      acc[0][j] = __builtin_amdgcn_wmma_f32_16x16x32_bf16(
          false, a0.v, false, b[j].v, (short)0, acc[0][j], false, false);
      acc[1][j] = __builtin_amdgcn_wmma_f32_16x16x32_bf16(
          false, a1.v, false, b[j].v, (short)0, acc[1][j], false, false);
    }
    __syncthreads();
  }

  // C layout: VGPR r -> M = r (lanes 0-15) / M = 8+r (lanes 16-31), N = lane&15
#pragma unroll
  for (int i = 0; i < 2; ++i) {
#pragma unroll
    for (int j = 0; j < 4; ++j) {
      const int n = c0 + j * 16 + l16;
      const float bv = bias[n];
      const int mb = r0 + i * 16 + hl * 8;
#pragma unroll
      for (int r = 0; r < 8; ++r) {
        float v = acc[i][j][r] + bv;
        if (RELU) v = fmaxf(v, 0.0f);
        if (OUT_BF16)
          ((u16*)Cout)[(size_t)(mb + r) * N + n] = f32_to_bf16(v);
        else
          ((float*)Cout)[(size_t)(mb + r) * N + n] = v;
      }
    }
  }
}

// ---------------- per-sample slogdet ----------------
// Block = 64 threads: wave 0 -> up spin, wave 1 -> dn spin. 32x32 LU with
// partial pivoting in LDS; lane = column for build/swap, lane = row for elim.
__global__ __launch_bounds__(64) void slogdet_kernel(
    const int* __restrict__ x, const float* __restrict__ orbitals,
    const float* __restrict__ corr, float* __restrict__ out)
{
  __shared__ float Amat[2][32][33];
  __shared__ int   idx[2][32];
  __shared__ float res[2][2];

  const int b = blockIdx.x;
  const int sp = threadIdx.x >> 5;    // 0 = up, 1 = dn
  const int lane = threadIdx.x & 31;

  if (lane == 0) {
    int cnt = 0;
    const int bit = sp + 1;           // up: bit0, dn: bit1
    for (int i = 0; i < 128 && cnt < 32; ++i)
      if (x[b * 128 + i] & bit) idx[sp][cnt++] = i;
  }
  __syncthreads();

  for (int i = 0; i < 32; ++i) {
    int o = idx[sp][i];
    Amat[sp][i][lane] = orbitals[o * 32 + lane] + corr[((size_t)b * 128 + o) * 32 + lane];
  }
  __syncthreads();

  float sign = 1.0f, logdet = 0.0f;
  for (int k = 0; k < 32; ++k) {
    // argmax |A[i][k]| over i >= k
    float bv = (lane >= k) ? fabsf(Amat[sp][lane][k]) : -1.0f;
    int bi = lane;
    for (int off = 16; off > 0; off >>= 1) {
      float ov = __shfl_down(bv, off);
      int oi = __shfl_down(bi, off);
      if (ov > bv) { bv = ov; bi = oi; }
    }
    int p = __shfl(bi, 0);
    if (p != k) {                      // wave-uniform branch; lane swaps column
      float tv = Amat[sp][k][lane];
      Amat[sp][k][lane] = Amat[sp][p][lane];
      Amat[sp][p][lane] = tv;
      sign = -sign;
    }
    __syncthreads();
    float pivot = Amat[sp][k][k];
    logdet += logf(fabsf(pivot));
    if (pivot < 0.0f) sign = -sign;
    if (lane > k) {                    // lane owns row `lane`
      float f = Amat[sp][lane][k] / pivot;
      for (int j = k + 1; j < 32; ++j)
        Amat[sp][lane][j] -= f * Amat[sp][k][j];
    }
    __syncthreads();
  }

  if (lane == 0) { res[sp][0] = logdet; res[sp][1] = sign; }
  __syncthreads();
  if (threadIdx.x == 0) {
    out[2 * b]     = res[0][0] + res[1][0];                 // Re: log|det_up| + log|det_dn|
    out[2 * b + 1] = (res[0][1] * res[1][1] < 0.0f)
                         ? 3.14159265358979323846f : 0.0f;  // Im: log(s_up*s_dn + 0j)
  }
}

// ---------------- launcher ----------------
extern "C" void kernel_launch(void* const* d_in, const int* in_sizes, int n_in,
                              void* d_out, int out_size, void* d_ws, size_t ws_size,
                              hipStream_t stream) {
  const int B = 1024, NORB = 128, HID = 4096;
  const int K1 = 4 * NORB;        // 512
  const int NOUT = NORB * 32;     // 4096

  const int*   x        = (const int*)d_in[0];
  const float* orbitals = (const float*)d_in[1];
  const float* W1       = (const float*)d_in[2];
  const float* b1       = (const float*)d_in[3];
  const float* W2       = (const float*)d_in[4];
  const float* b2       = (const float*)d_in[5];
  float* out = (float*)d_out;

  char* p = (char*)d_ws;
  u16* onehot = (u16*)p;   p += (size_t)B * K1 * 2;      //  1 MiB
  u16* W1t    = (u16*)p;   p += (size_t)HID * K1 * 2;    //  4 MiB
  u16* h      = (u16*)p;   p += (size_t)B * HID * 2;     //  8 MiB
  u16* W2t    = (u16*)p;   p += (size_t)NOUT * HID * 2;  // 32 MiB
  float* corr = (float*)p; p += (size_t)B * NOUT * 4;    // 16 MiB

  onehot_kernel<<<(B * NORB + 255) / 256, 256, 0, stream>>>(x, onehot, B * NORB);
  transpose_to_bf16<<<dim3(HID / 32, K1 / 32), dim3(32, 8), 0, stream>>>(
      W1, W1t, K1, HID);
  transpose_to_bf16<<<dim3(NOUT / 32, HID / 32), dim3(32, 8), 0, stream>>>(
      W2, W2t, HID, NOUT);

  gemm_bf16<1, 1><<<dim3(HID / 256, B / 64), 256, 0, stream>>>(
      onehot, W1t, b1, (void*)h, B, HID, K1);
  gemm_bf16<0, 0><<<dim3(NOUT / 256, B / 64), 256, 0, stream>>>(
      h, W2t, b2, (void*)corr, B, NOUT, HID);

  slogdet_kernel<<<B, 64, 0, stream>>>(x, orbitals, corr, out);
}
